// LinkEncoding_18708877541735
// MI455X (gfx1250) — matrix operand order repaired and checked by reading, not compiled, not run on hardware
//
#include <hip/hip_runtime.h>
#include <hip/hip_bf16.h>
#include <math.h>

#define EDGES   800000
#define NNODES  50000
#define KTOK    30
#define CIN_P   224      // 200 (TCH+IN_CH) padded to 7*32
#define HID_P   112      // 100 padded to 7*16
#define HID_P2  128      // 100 padded to 4*32 (A-fragment K dim)
#define FF      400
#define FF_P    416      // 400 padded to 13*32
#define MIXER_WAVE_BYTES 39424

typedef __attribute__((ext_vector_type(16))) _Float16 v16h;
typedef __attribute__((ext_vector_type(8)))  _Float16 v8h;
typedef __attribute__((ext_vector_type(8)))  float    v8f;

static __device__ __forceinline__ int lane_id() { return threadIdx.x & 31; }

static __device__ __forceinline__ v16h make_frag(v8h lo, v8h hi) {
  v16h a;
#pragma unroll
  for (int j = 0; j < 8; ++j) { a[j] = lo[j]; a[8 + j] = hi[j]; }
  return a;
}

// A fragment (16x32 f16) from row-major A[16][ld]; per ISA layout:
// lane L: row m=L&15, elems j=0..7 -> K=(L>>4)*8+j, j=8..15 -> K=16+(L>>4)*8+(j-8)
static __device__ __forceinline__ v16h load_a_frag(const _Float16* A, int ld) {
  int l = lane_id(); int m = l & 15; int hs = l >> 4;
  const _Float16* p = A + m * ld + hs * 8;
  return make_frag(*(const v8h*)p, *(const v8h*)(p + 16));
}

// B fragment (32x16 f16) from W[N][K] row-major (= B^T, PyTorch weight layout):
// lane L: col n=L&15, elems j=0..7 -> K=(L>>4)*8+j, j=8..15 -> K=16+(L>>4)*8+(j-8)
static __device__ __forceinline__ v16h load_b_frag(const _Float16* W, int ld) {
  int l = lane_id(); int n = l & 15; int ks = l >> 4;
  const _Float16* p = W + n * ld + ks * 8;
  return make_frag(*(const v8h*)p, *(const v8h*)(p + 16));
}

static __device__ __forceinline__ v8f wmma_f16(v16h a, v16h b, v8f c) {
  return __builtin_amdgcn_wmma_f32_16x16x32_f16(false, a, false, b, (short)0, c,
                                                false, false);
}

static __device__ __forceinline__ float gelu_exact(float x) {
  return 0.5f * x * (1.0f + erff(x * 0.70710678118654752f));
}

// Async global -> LDS 16-byte copy (CDNA5, tracked by ASYNCcnt)
static __device__ __forceinline__ void async_g2l_b128(unsigned lds_byte_off,
                                                      const void* gaddr) {
  asm volatile("global_load_async_to_lds_b128 %0, %1, off"
               :: "v"(lds_byte_off), "v"(gaddr) : "memory");
}
static __device__ __forceinline__ void wait_asynccnt0() {
  asm volatile("s_wait_asynccnt 0x0" ::: "memory");
}

// ---------------- kernel 0: pack f32 weights -> padded f16 [padN][padK] -----
__global__ void pack_w_f16(_Float16* __restrict__ dst, const float* __restrict__ src,
                           int realN, int realK, int padK, int total) {
  for (int i = blockIdx.x * blockDim.x + threadIdx.x; i < total;
       i += gridDim.x * blockDim.x) {
    int n = i / padK, k = i % padK;
    float v = (n < realN && k < realK) ? src[n * realK + k] : 0.0f;
    dst[i] = (_Float16)v;
  }
}

// ---------------- kernel 1: head GEMM  h[E,112] = [cos(t*tw), attr] @ Wh^T ---
__global__ void head_kernel(const float* __restrict__ edge_attr,
                            const float* __restrict__ edge_time,
                            const _Float16* __restrict__ Wh,   // [112][224]
                            const float* __restrict__ head_b,
                            _Float16* __restrict__ h_out) {    // [E][112]
  extern __shared__ char smem_raw[];
  _Float16* smem = (_Float16*)smem_raw;
  int wave = threadIdx.x >> 5;
  int l = threadIdx.x & 31;
  _Float16* A = smem + wave * 16 * CIN_P;
  int ebase = blockIdx.x * 128 + wave * 16;
  __builtin_prefetch(edge_attr + (size_t)ebase * 100, 0, 1);
  const float C2 = -0.33554829f;  // -(10/99)*log2(10) : tw[c] = 2^(c*C2)
  for (int i = l; i < 16 * CIN_P; i += 32) {
    int r = i / CIN_P, c = i % CIN_P;
    int e = ebase + r;
    float v = 0.0f;
    if (c < 100)      v = cosf(edge_time[e] * exp2f((float)c * C2));
    else if (c < 200) v = edge_attr[(size_t)e * 100 + (c - 100)];
    A[i] = (_Float16)v;
  }
  __syncthreads();
#pragma unroll
  for (int nt = 0; nt < 7; ++nt) {
    v8f acc = {};
#pragma unroll
    for (int kc = 0; kc < 7; ++kc) {
      v16h a = load_a_frag(A + kc * 32, CIN_P);
      v16h b = load_b_frag(Wh + nt * 16 * CIN_P + kc * 32, CIN_P);
      acc = wmma_f16(a, b, acc);
    }
    int n = nt * 16 + (l & 15);
    float bias = (n < 100) ? head_b[n] : 0.0f;
#pragma unroll
    for (int r = 0; r < 8; ++r) {
      int m = r + 8 * (l >> 4);
      int e = ebase + m;
      h_out[(size_t)e * HID_P + n] = (_Float16)((n < 100) ? (acc[r] + bias) : 0.0f);
    }
  }
}

// ---------------- kernel 2a/2b: per-node first-K edge slots ------------------
__global__ void init_slots(int* __restrict__ cnt, int* __restrict__ slots,
                           _Float16* __restrict__ zrow) {
  int i = blockIdx.x * blockDim.x + threadIdx.x;
  if (i < NNODES) cnt[i] = 0;
  if (i < 128) zrow[i] = (_Float16)0.0f;   // 256B zero source for async gather
  for (int j = i; j < NNODES * KTOK; j += gridDim.x * blockDim.x) slots[j] = -1;
}

__global__ void scatter_kernel(const int* __restrict__ node_batch,
                               int* __restrict__ cnt, int* __restrict__ slots) {
  int e = blockIdx.x * blockDim.x + threadIdx.x;
  if (e >= EDGES) return;
  int nd = node_batch[e];
  int r = atomicAdd(&cnt[nd], 1);
  if (r < KTOK) slots[nd * KTOK + r] = e;
}

// ---------------- kernel 3: fused per-node MLP-mixer -------------------------
// 128 threads = 4 waves, one node per wave.
__global__ void mixer_kernel(const _Float16* __restrict__ h_f16,
                             const int* __restrict__ slots,
                             const _Float16* __restrict__ zrow,
                             const _Float16* __restrict__ Wt1,  // [16][32]
                             const _Float16* __restrict__ Wt2,  // [32][32]
                             const _Float16* __restrict__ Wc1,  // [416][128]
                             const _Float16* __restrict__ Wc2,  // [112][416]
                             const float* __restrict__ ln_t_g, const float* __restrict__ ln_t_b,
                             const float* __restrict__ tok1_b, const float* __restrict__ tok2_b,
                             const float* __restrict__ ln_c_g, const float* __restrict__ ln_c_b,
                             const float* __restrict__ ch1_b,  const float* __restrict__ ch2_b,
                             const float* __restrict__ ln_h_g, const float* __restrict__ ln_h_b,
                             _Float16* __restrict__ Tbuf) {     // [N][128]
  extern __shared__ char smem_raw[];
  int wave = threadIdx.x >> 5;
  int l = threadIdx.x & 31;
  char* wm = smem_raw + wave * MIXER_WAVE_BYTES;
  _Float16* xraw = (_Float16*)wm;          // [32][112] gathered tokens
  _Float16* xt   = xraw + 32 * 112;        // [112][32] LN(x)^T; reused as hch
  _Float16* tmid = xt + 112 * 32;          // [112][32] gelu token-mid
  _Float16* hbuf = tmid + 112 * 32;        // [32][112] h_token
  _Float16* xbuf = hbuf + 32 * 112;        // [32][128] LN(h_token)
  _Float16* ubuf = xbuf + 32 * 128;        // [32][32]  gelu(ch1) staging
  float*    tvec = (float*)(ubuf + 32 * 32);  // [112] f32 accum

  int node = blockIdx.x * 4 + wave;

  // ---- async gather x[32][112] from h_f16 (zeros sourced from zrow) ----
  // 32 rows x 14 chunks of 16B; ASYNCcnt-tracked global->LDS DMA.
  {
    unsigned xoff = (unsigned)(uintptr_t)(void*)xraw;
#pragma unroll
    for (int it = 0; it < 14; ++it) {
      int idx = it * 32 + l;
      int r = idx / 14, cb = idx % 14;
      const void* g;
      if (r < KTOK) {
        int e = slots[node * KTOK + r];
        g = (e >= 0) ? (const void*)(h_f16 + (size_t)e * HID_P + cb * 8)
                     : (const void*)(zrow + cb * 8);
      } else {
        g = (const void*)(zrow + cb * 8);
      }
      async_g2l_b128(xoff + (unsigned)(r * 112 + cb * 8) * 2, g);
    }
  }
  for (int i = l; i < 112; i += 32) tvec[i] = 0.0f;
  wait_asynccnt0();
  __syncthreads();

  // ---- LN over channels (per token row) -> transposed xt[c][r] ----
  {
    int r = l;
    if (r < KTOK) {
      float m = 0.f, v2 = 0.f;
      for (int c = 0; c < 100; ++c) m += (float)xraw[r * 112 + c];
      m *= 0.01f;
      for (int c = 0; c < 100; ++c) { float d = (float)xraw[r * 112 + c] - m; v2 += d * d; }
      float rs = rsqrtf(v2 * 0.01f + 1e-5f);
      for (int c = 0; c < 100; ++c)
        xt[c * 32 + r] = (_Float16)(((float)xraw[r * 112 + c] - m) * rs * ln_t_g[c] + ln_t_b[c]);
      for (int c = 100; c < 112; ++c) xt[c * 32 + r] = (_Float16)0.f;
    } else {
      for (int c = 0; c < 112; ++c) xt[c * 32 + r] = (_Float16)0.f;
    }
  }
  __syncthreads();

  // ---- token matmul 1: [112ch x 32tok] @ Wt1^T -> gelu -> tmid[112][32] ----
  {
    v16h bt1 = load_b_frag(Wt1, 32);
    int n = l & 15;
    float bias = (n < 15) ? tok1_b[n] : 0.f;
#pragma unroll
    for (int mt = 0; mt < 7; ++mt) {
      v16h a = load_a_frag(xt + mt * 16 * 32, 32);
      v8f c = {};
      c = wmma_f16(a, bt1, c);
#pragma unroll
      for (int r = 0; r < 8; ++r) {
        int ch = mt * 16 + r + 8 * (l >> 4);
        tmid[ch * 32 + n] = (_Float16)gelu_exact(c[r] + bias);
        tmid[ch * 32 + n + 16] = (_Float16)0.f;   // zero K-pad columns
      }
    }
  }
  __syncthreads();

  // ---- token matmul 2 + residual -> hbuf[tok][ch] ----
#pragma unroll
  for (int nt = 0; nt < 2; ++nt) {
    v16h b = load_b_frag(Wt2 + nt * 16 * 32, 32);
    int tok = nt * 16 + (l & 15);
    float bias = (tok < KTOK) ? tok2_b[tok] : 0.f;
#pragma unroll
    for (int mt = 0; mt < 7; ++mt) {
      v16h a = load_a_frag(tmid + mt * 16 * 32, 32);
      v8f c = {};
      c = wmma_f16(a, b, c);
#pragma unroll
      for (int r = 0; r < 8; ++r) {
        int ch = mt * 16 + r + 8 * (l >> 4);
        float ht = c[r] + bias + (float)xraw[tok * 112 + ch];
        hbuf[tok * 112 + ch] = (_Float16)((tok < KTOK) ? ht : 0.f);
      }
    }
  }
  __syncthreads();

  // ---- LN over channels -> xbuf[32][128] (K-padded for A frags) ----
  {
    int r = l;
    if (r < KTOK) {
      float m = 0.f, v2 = 0.f;
      for (int c = 0; c < 100; ++c) m += (float)hbuf[r * 112 + c];
      m *= 0.01f;
      for (int c = 0; c < 100; ++c) { float d = (float)hbuf[r * 112 + c] - m; v2 += d * d; }
      float rs = rsqrtf(v2 * 0.01f + 1e-5f);
      for (int c = 0; c < 100; ++c)
        xbuf[r * 128 + c] = (_Float16)(((float)hbuf[r * 112 + c] - m) * rs * ln_c_g[c] + ln_c_b[c]);
      for (int c = 100; c < 128; ++c) xbuf[r * 128 + c] = (_Float16)0.f;
    } else {
      for (int c = 0; c < 128; ++c) xbuf[r * 128 + c] = (_Float16)0.f;
    }
  }
  __syncthreads();

  // ---- fused channel MLP: gelu(xbuf @ Wc1^T) @ Wc2^T accumulated in regs ----
  v16h afr[2][4];
#pragma unroll
  for (int mt = 0; mt < 2; ++mt)
#pragma unroll
    for (int kc = 0; kc < 4; ++kc)
      afr[mt][kc] = load_a_frag(xbuf + mt * 16 * 128 + kc * 32, 128);

  v8f acc2[2][7];
#pragma unroll
  for (int mt = 0; mt < 2; ++mt)
#pragma unroll
    for (int nt = 0; nt < 7; ++nt) { v8f z = {}; acc2[mt][nt] = z; }

  for (int jc = 0; jc < 13; ++jc) {
#pragma unroll
    for (int sub = 0; sub < 2; ++sub) {
      int nbase = jc * 32 + sub * 16;
      v8f u0 = {}, u1 = {};
#pragma unroll
      for (int kc = 0; kc < 4; ++kc) {
        v16h b = load_b_frag(Wc1 + (size_t)nbase * HID_P2 + kc * 32, HID_P2);
        u0 = wmma_f16(afr[0][kc], b, u0);
        u1 = wmma_f16(afr[1][kc], b, u1);
      }
      int n = nbase + (l & 15);
      float bias = (n < FF) ? ch1_b[n] : 0.f;
#pragma unroll
      for (int r = 0; r < 8; ++r) {
        int tok0 = r + 8 * (l >> 4);
        ubuf[tok0 * 32 + sub * 16 + (l & 15)] = (_Float16)gelu_exact(u0[r] + bias);
        ubuf[(tok0 + 16) * 32 + sub * 16 + (l & 15)] = (_Float16)gelu_exact(u1[r] + bias);
      }
    }
    __syncthreads();
    v16h a0 = load_a_frag(ubuf, 32);
    v16h a1 = load_a_frag(ubuf + 16 * 32, 32);
#pragma unroll
    for (int nt = 0; nt < 7; ++nt) {
      v16h b = load_b_frag(Wc2 + (size_t)nt * 16 * FF_P + jc * 32, FF_P);
      acc2[0][nt] = wmma_f16(a0, b, acc2[0][nt]);
      acc2[1][nt] = wmma_f16(a1, b, acc2[1][nt]);
    }
    __syncthreads();
  }

  // ---- h_channel = acc2 + ch2_b + h_token -> hch (reuse xt region) ----
  _Float16* hch = xt;
#pragma unroll
  for (int mt = 0; mt < 2; ++mt)
#pragma unroll
    for (int nt = 0; nt < 7; ++nt) {
      int ch = nt * 16 + (l & 15);
      float bias = (ch < 100) ? ch2_b[ch] : 0.f;
#pragma unroll
      for (int r = 0; r < 8; ++r) {
        int tok = mt * 16 + r + 8 * (l >> 4);
        float v = acc2[mt][nt][r] + bias + (float)hbuf[tok * 112 + ch];
        hch[tok * 112 + ch] = (_Float16)v;
      }
    }
  __syncthreads();

  // ---- final LN + mean over K (ds f32 atomics) -> Tbuf[node][128] ----
  {
    int r = l;
    if (r < KTOK) {
      float m = 0.f, v2 = 0.f;
      for (int c = 0; c < 100; ++c) m += (float)hch[r * 112 + c];
      m *= 0.01f;
      for (int c = 0; c < 100; ++c) { float d = (float)hch[r * 112 + c] - m; v2 += d * d; }
      float rs = rsqrtf(v2 * 0.01f + 1e-5f);
      for (int c = 0; c < 100; ++c) {
        float t = ((float)hch[r * 112 + c] - m) * rs * ln_h_g[c] + ln_h_b[c];
        atomicAdd(&tvec[c], t);
      }
    }
  }
  __syncthreads();
  const float inv_k = 1.0f / 30.0f;
  for (int c = l; c < 128; c += 32) {
    float v = (c < 100) ? tvec[c] * inv_k : 0.f;
    Tbuf[(size_t)node * 128 + c] = (_Float16)v;
  }
}

// ---------------- kernel 4: output GEMM  out[N,100] = T @ out_w^T + b --------
__global__ void out_kernel(const _Float16* __restrict__ Tbuf,  // [N][128]
                           const _Float16* __restrict__ Wo,    // [112][128]
                           const float* __restrict__ out_b,
                           float* __restrict__ out) {          // [N][100]
  int wave = threadIdx.x >> 5;
  int l = threadIdx.x & 31;
  int mtile = blockIdx.x * 8 + wave;
  if (mtile >= NNODES / 16) return;
  const _Float16* A = Tbuf + (size_t)mtile * 16 * 128;
  v16h afr[4];
#pragma unroll
  for (int kc = 0; kc < 4; ++kc) afr[kc] = load_a_frag(A + kc * 32, 128);
#pragma unroll
  for (int nt = 0; nt < 7; ++nt) {
    v8f acc = {};
#pragma unroll
    for (int kc = 0; kc < 4; ++kc) {
      v16h b = load_b_frag(Wo + nt * 16 * 128 + kc * 32, 128);
      acc = wmma_f16(afr[kc], b, acc);
    }
    int n = nt * 16 + (l & 15);
    if (n < 100) {
      float bias = out_b[n];
#pragma unroll
      for (int r = 0; r < 8; ++r) {
        int m = mtile * 16 + r + 8 * (l >> 4);
        out[(size_t)m * 100 + n] = acc[r] + bias;
      }
    }
  }
}

// ---------------- launch -----------------------------------------------------
extern "C" void kernel_launch(void* const* d_in, const int* in_sizes, int n_in,
                              void* d_out, int out_size, void* d_ws, size_t ws_size,
                              hipStream_t stream) {
  (void)in_sizes; (void)n_in; (void)out_size; (void)ws_size;
  const float* edge_attr = (const float*)d_in[0];
  const float* edge_time = (const float*)d_in[1];
  const int*   node_batch = (const int*)d_in[2];
  const float* head_w = (const float*)d_in[4];
  const float* head_b = (const float*)d_in[5];
  const float* ln_t_g = (const float*)d_in[6];
  const float* ln_t_b = (const float*)d_in[7];
  const float* tok1_w = (const float*)d_in[8];
  const float* tok1_b = (const float*)d_in[9];
  const float* tok2_w = (const float*)d_in[10];
  const float* tok2_b = (const float*)d_in[11];
  const float* ln_c_g = (const float*)d_in[12];
  const float* ln_c_b = (const float*)d_in[13];
  const float* ch1_w = (const float*)d_in[14];
  const float* ch1_b = (const float*)d_in[15];
  const float* ch2_w = (const float*)d_in[16];
  const float* ch2_b = (const float*)d_in[17];
  const float* ln_h_g = (const float*)d_in[18];
  const float* ln_h_b = (const float*)d_in[19];
  const float* out_w = (const float*)d_in[20];
  const float* out_b = (const float*)d_in[21];
  float* out = (float*)d_out;

  char* ws = (char*)d_ws;
  size_t off = 0;
  auto carve = [&](size_t bytes) -> char* {
    char* p = ws + off;
    off += (bytes + 255) & ~(size_t)255;
    return p;
  };
  _Float16* h_f16 = (_Float16*)carve((size_t)EDGES * HID_P * 2);
  _Float16* Tbuf  = (_Float16*)carve((size_t)NNODES * 128 * 2);
  int* slots = (int*)carve((size_t)NNODES * KTOK * 4);
  int* cnt   = (int*)carve((size_t)NNODES * 4);
  _Float16* zrow = (_Float16*)carve(256);
  _Float16* Wh  = (_Float16*)carve((size_t)HID_P * CIN_P * 2);
  _Float16* Wt1 = (_Float16*)carve(16 * 32 * 2);
  _Float16* Wt2 = (_Float16*)carve(32 * 32 * 2);
  _Float16* Wc1 = (_Float16*)carve((size_t)FF_P * HID_P2 * 2);
  _Float16* Wc2 = (_Float16*)carve((size_t)HID_P * FF_P * 2);
  _Float16* Wo  = (_Float16*)carve((size_t)HID_P * HID_P2 * 2);

  auto blk = [](int total) { return (total + 255) / 256; };
  pack_w_f16<<<blk(HID_P * CIN_P), 256, 0, stream>>>(Wh, head_w, 100, 200, CIN_P, HID_P * CIN_P);
  pack_w_f16<<<blk(16 * 32), 256, 0, stream>>>(Wt1, tok1_w, 15, 30, 32, 16 * 32);
  pack_w_f16<<<blk(32 * 32), 256, 0, stream>>>(Wt2, tok2_w, 30, 15, 32, 32 * 32);
  pack_w_f16<<<blk(FF_P * HID_P2), 256, 0, stream>>>(Wc1, ch1_w, 400, 100, HID_P2, FF_P * HID_P2);
  pack_w_f16<<<blk(HID_P * FF_P), 256, 0, stream>>>(Wc2, ch2_w, 100, 400, FF_P, HID_P * FF_P);
  pack_w_f16<<<blk(HID_P * HID_P2), 256, 0, stream>>>(Wo, out_w, 100, 100, HID_P2, HID_P * HID_P2);

  init_slots<<<blk(NNODES * KTOK), 256, 0, stream>>>(cnt, slots, zrow);

  head_kernel<<<EDGES / 128, 256, 8 * 16 * CIN_P * 2, stream>>>(
      edge_attr, edge_time, Wh, head_b, h_f16);

  scatter_kernel<<<blk(EDGES), 256, 0, stream>>>(node_batch, cnt, slots);

  mixer_kernel<<<NNODES / 4, 128, 4 * MIXER_WAVE_BYTES, stream>>>(
      h_f16, slots, zrow, Wt1, Wt2, Wc1, Wc2,
      ln_t_g, ln_t_b, tok1_b, tok2_b, ln_c_g, ln_c_b,
      ch1_b, ch2_b, ln_h_g, ln_h_b, Tbuf);

  out_kernel<<<((NNODES / 16) + 7) / 8, 256, 0, stream>>>(Tbuf, Wo, out_b, out);
}
